// RULSTM_30064771072271
// MI455X (gfx1250) — matrix operand order repaired and checked by reading, not compiled, not run on hardware
//
#include <hip/hip_runtime.h>
#include <stdint.h>

// RULSTM for MI455X (gfx1250): bf16 WMMA GEMMs (v_wmma_f32_16x16x32_bf16),
// f32 accumulate, fused in-register LSTM gates, t-batched unroll phase,
// double-buffered K loop (loads for tile k+32 in flight during WMMAs of tile k).

#define Bb 128
#define Tt 16
#define Ff 1024
#define Hh 1024
#define Cc 2513
#define CPAD 2560

typedef __attribute__((ext_vector_type(16))) __bf16 v16bf;
typedef __attribute__((ext_vector_type(8)))  float  v8f;

union FragBF { v16bf v; uint4 q[2]; };

__device__ __forceinline__ unsigned short f2bf(float f) {
  union { float f; uint32_t u; } x; x.f = f;
  uint32_t u = x.u;
  if ((u & 0x7F800000u) == 0x7F800000u) return (unsigned short)(u >> 16); // inf/nan: truncate
  return (unsigned short)((u + 0x7FFFu + ((u >> 16) & 1u)) >> 16);        // RNE
}

// A fragment (16x32 bf16): lane holds row lane%16, two 8-elem K-chunks at
// k0 = 8*(lane/16) and k0+16  ->  two b128 loads from row-major K-contiguous A.
template <int N>
__device__ __forceinline__ void load_frags_A(FragBF (&f)[N], const unsigned short* base,
                                             int r0, int ld, int kk, int lr, int lh) {
  #pragma unroll
  for (int i = 0; i < N; ++i) {
    const unsigned short* p = base + (size_t)(r0 + 16 * i + lr) * ld + kk + 8 * lh;
    f[i].q[0] = *(const uint4*)p;
    f[i].q[1] = *(const uint4*)(p + 16);
  }
}

// B fragment (32x16 bf16): lane holds column lane%16, two 8-elem K-chunks at
// k0 = 8*(lane/16) and k0+16  ->  two b128 loads from row-major [N,K] weights.
template <int N>
__device__ __forceinline__ void load_frags_B(FragBF (&f)[N], const unsigned short* base,
                                             int r0, int rstride, int ld, int kk, int lr, int lh) {
  #pragma unroll
  for (int i = 0; i < N; ++i) {
    const unsigned short* p = base + (size_t)(r0 + rstride * i + lr) * ld + kk + 8 * lh;
    f[i].q[0] = *(const uint4*)p;
    f[i].q[1] = *(const uint4*)(p + 16);
  }
}

// ---------------- conversion / small elementwise kernels ----------------

__global__ void cvt_bf16(const float* __restrict__ in, unsigned short* __restrict__ out, int n) {
  int i = blockIdx.x * blockDim.x + threadIdx.x;
  if (i < n) out[i] = f2bf(in[i]);
}

// inputs [B,T,F] -> X_b [T,B,F] bf16 (the reference's transpose(1,0,2))
__global__ void cvt_x_tbf(const float* __restrict__ in, unsigned short* __restrict__ out, int n) {
  int i = blockIdx.x * blockDim.x + threadIdx.x;
  if (i >= n) return;
  int k = i & (Ff - 1);
  int r = i >> 10;           // t*B + b
  int b = r & (Bb - 1);
  int t = r >> 7;
  out[i] = f2bf(in[((size_t)b * Tt + t) * Ff + k]);
}

__global__ void add_vec(const float* __restrict__ a, const float* __restrict__ b,
                        float* __restrict__ o, int n) {
  int i = blockIdx.x * blockDim.x + threadIdx.x;
  if (i < n) o[i] = a[i] + b[i];
}

// ---------------- generic bf16 GEMM: C[M,Nout] = A[M,K] * Bw[N,K]^T + bias ----------------
// Block: 256 threads = 8 waves; wave tile 64x32 (4x2 WMMA tiles); block tile 128x128.
__global__ __launch_bounds__(256) void gemm_bf16_bias(
    const unsigned short* __restrict__ A,   // [M,K] bf16 row-major
    const unsigned short* __restrict__ Bw,  // [Nalloc,K] bf16 row-major
    const float* __restrict__ bias,         // [Nout]
    float* __restrict__ C,
    int Nout, int K, int remap)             // remap: out_row = b*T + t for m = t*B + b
{
  int lane = threadIdx.x & 31;
  int wave = threadIdx.x >> 5;
  int lr = lane & 15, lh = lane >> 4;
  int m0 = blockIdx.y * 128 + (wave >> 2) * 64;
  int n0 = blockIdx.x * 128 + (wave & 3) * 32;

  v8f acc[4][2];
  #pragma unroll
  for (int i = 0; i < 4; ++i)
    #pragma unroll
    for (int j = 0; j < 2; ++j)
      #pragma unroll
      for (int e = 0; e < 8; ++e) acc[i][j][e] = 0.f;

  FragBF a0[4], b0[2], a1[4], b1[2];
  load_frags_A(a0, A, m0, K, 0, lr, lh);
  load_frags_B(b0, Bw, n0, 16, K, 0, lr, lh);

  for (int kk = 0; kk < K; kk += 64) {   // K is a multiple of 64
    load_frags_A(a1, A, m0, K, kk + 32, lr, lh);
    load_frags_B(b1, Bw, n0, 16, K, kk + 32, lr, lh);
    #pragma unroll
    for (int ms = 0; ms < 4; ++ms)
      #pragma unroll
      for (int ns = 0; ns < 2; ++ns)
        acc[ms][ns] = __builtin_amdgcn_wmma_f32_16x16x32_bf16(
            false, a0[ms].v, false, b0[ns].v, (short)0, acc[ms][ns], false, false);
    if (kk + 64 < K) {
      load_frags_A(a0, A, m0, K, kk + 64, lr, lh);
      load_frags_B(b0, Bw, n0, 16, K, kk + 64, lr, lh);
    }
    #pragma unroll
    for (int ms = 0; ms < 4; ++ms)
      #pragma unroll
      for (int ns = 0; ns < 2; ++ns)
        acc[ms][ns] = __builtin_amdgcn_wmma_f32_16x16x32_bf16(
            false, a1[ms].v, false, b1[ns].v, (short)0, acc[ms][ns], false, false);
  }

  #pragma unroll
  for (int ms = 0; ms < 4; ++ms)
    #pragma unroll
    for (int ns = 0; ns < 2; ++ns)
      #pragma unroll
      for (int v = 0; v < 8; ++v) {
        int m = m0 + 16 * ms + 8 * lh + v;
        int n = n0 + 16 * ns + lr;
        if (n < Nout) {
          int orow = remap ? ((m & (Bb - 1)) * Tt + (m >> 7)) : m;
          C[(size_t)orow * Nout + n] = acc[ms][ns][v] + bias[n];
        }
      }
}

// ---------------- fused LSTM recurrent step ----------------
// pre = xi[t] + h_src[t] @ Wh^T ; gates -> (h_dst[t], c_dst[t]); finals to h_fin.
// Block: 256 thr = 8 waves; wave computes one 32x16 (row,col) tile for ALL 4 gates
// (columns n, H+n, 2H+n, 3H+n) -> gate math entirely in registers, no LDS/barrier.
// grid = (H/32, active_t)
__global__ __launch_bounds__(256) void lstm_step(
    const unsigned short* __restrict__ Wh,    // [4H,H] bf16
    const unsigned short* __restrict__ h_src, // [gy,B,H] bf16
    const float* __restrict__ xi,             // [gy,B,4H] f32
    const float* c_src,                       // [gy,B,H] f32 (may == c_dst)
    float* c_dst,
    unsigned short* __restrict__ h_dst,       // [gy,B,H] bf16
    unsigned short* h_fin,                    // [T,B,H] bf16 or null
    int t_last)
{
  int t = blockIdx.y;
  const unsigned short* hsp = h_src + (size_t)t * Bb * Hh;
  const float* xip = xi + (size_t)t * Bb * 4 * Hh;
  const float* csp = c_src + (size_t)t * Bb * Hh;
  float* cdp = c_dst + (size_t)t * Bb * Hh;
  unsigned short* hdp = h_dst + (size_t)t * Bb * Hh;

  int lane = threadIdx.x & 31;
  int wave = threadIdx.x >> 5;
  int lr = lane & 15, lh = lane >> 4;
  int m0 = (wave & 3) * 32;                        // batch rows [m0, m0+32)
  int col0 = blockIdx.x * 32 + (wave >> 2) * 16;   // gate column [col0, col0+16)

  v8f acc[4][2];   // [gate][m-subtile]
  #pragma unroll
  for (int g = 0; g < 4; ++g)
    #pragma unroll
    for (int j = 0; j < 2; ++j)
      #pragma unroll
      for (int e = 0; e < 8; ++e) acc[g][j][e] = 0.f;

  FragBF a0[2], b0[4], a1[2], b1[4];
  load_frags_A(a0, hsp, m0, Hh, 0, lr, lh);
  load_frags_B(b0, Wh, col0, Hh, Hh, 0, lr, lh);   // rows g*H + col0, g = 0..3

  for (int kk = 0; kk < Hh; kk += 64) {
    load_frags_A(a1, hsp, m0, Hh, kk + 32, lr, lh);
    load_frags_B(b1, Wh, col0, Hh, Hh, kk + 32, lr, lh);
    #pragma unroll
    for (int g = 0; g < 4; ++g)
      #pragma unroll
      for (int ms = 0; ms < 2; ++ms)
        acc[g][ms] = __builtin_amdgcn_wmma_f32_16x16x32_bf16(
            false, a0[ms].v, false, b0[g].v, (short)0, acc[g][ms], false, false);
    if (kk + 64 < Hh) {
      load_frags_A(a0, hsp, m0, Hh, kk + 64, lr, lh);
      load_frags_B(b0, Wh, col0, Hh, Hh, kk + 64, lr, lh);
    }
    #pragma unroll
    for (int g = 0; g < 4; ++g)
      #pragma unroll
      for (int ms = 0; ms < 2; ++ms)
        acc[g][ms] = __builtin_amdgcn_wmma_f32_16x16x32_bf16(
            false, a1[ms].v, false, b1[g].v, (short)0, acc[g][ms], false, false);
  }

  bool fin = (h_fin != nullptr) && (t == t_last);
  #pragma unroll
  for (int ms = 0; ms < 2; ++ms)
    #pragma unroll
    for (int v = 0; v < 8; ++v) {
      int row = m0 + 16 * ms + 8 * lh + v;
      int col = col0 + lr;
      const float* xr = xip + (size_t)row * 4 * Hh + col;
      float ip = acc[0][ms][v] + xr[0];
      float fp = acc[1][ms][v] + xr[Hh];
      float gp = acc[2][ms][v] + xr[2 * Hh];
      float op = acc[3][ms][v] + xr[3 * Hh];
      float co = csp[(size_t)row * Hh + col];
      float si = 1.f / (1.f + __expf(-ip));
      float sf = 1.f / (1.f + __expf(-fp));
      float so = 1.f / (1.f + __expf(-op));
      float cn = sf * co + si * tanhf(gp);
      float hn = so * tanhf(cn);
      cdp[(size_t)row * Hh + col] = cn;
      unsigned short hb = f2bf(hn);
      hdp[(size_t)row * Hh + col] = hb;
      if (fin) h_fin[(size_t)t * Bb * Hh + (size_t)row * Hh + col] = hb;
    }
}

// ---------------- host orchestration ----------------

extern "C" void kernel_launch(void* const* d_in, const int* in_sizes, int n_in,
                              void* d_out, int out_size, void* d_ws, size_t ws_size,
                              hipStream_t stream) {
  const float* inputs = (const float*)d_in[0];
  const float* Wi_r = (const float*)d_in[1];
  const float* Wh_r = (const float*)d_in[2];
  const float* bi_r = (const float*)d_in[3];
  const float* bh_r = (const float*)d_in[4];
  const float* Wi_u = (const float*)d_in[5];
  const float* Wh_u = (const float*)d_in[6];
  const float* bi_u = (const float*)d_in[7];
  const float* bh_u = (const float*)d_in[8];
  const float* Wc   = (const float*)d_in[9];
  const float* bc   = (const float*)d_in[10];
  float* out = (float*)d_out;

  const size_t W4 = (size_t)4 * Hh * Ff;      // 4M elems
  const size_t BH = (size_t)Bb * Hh;
  char* p = (char*)d_ws;
  auto alloc = [&](size_t bytes) { char* r = p; p += (bytes + 255) & ~(size_t)255; return r; };

  unsigned short* Wi_r_b = (unsigned short*)alloc(W4 * 2);
  unsigned short* Wh_r_b = (unsigned short*)alloc(W4 * 2);
  unsigned short* Wi_u_b = (unsigned short*)alloc(W4 * 2);
  unsigned short* Wh_u_b = (unsigned short*)alloc(W4 * 2);
  unsigned short* Wc_b   = (unsigned short*)alloc((size_t)CPAD * Hh * 2);
  unsigned short* X_b    = (unsigned short*)alloc((size_t)Tt * Bb * Ff * 2);
  float* br   = (float*)alloc((size_t)4 * Hh * 4);
  float* bu   = (float*)alloc((size_t)4 * Hh * 4);
  float* XI_r = (float*)alloc((size_t)Tt * Bb * 4 * Hh * 4);
  float* XI_u = (float*)alloc((size_t)Tt * Bb * 4 * Hh * 4);
  unsigned short* hs = (unsigned short*)alloc((size_t)Tt * BH * 2);  // roll out / unroll ping
  unsigned short* hp = (unsigned short*)alloc((size_t)Tt * BH * 2);  // unroll pong
  unsigned short* hf = (unsigned short*)alloc((size_t)Tt * BH * 2);  // final hiddens h_n
  float* cs = (float*)alloc((size_t)Tt * BH * 4);                    // cell state (in place)
  unsigned short* h0 = (unsigned short*)alloc(BH * 2);               // zero h init
  float* c0 = (float*)alloc(BH * 4);                                 // zero c init

  dim3 blk(256);
  int n;

  // weights / inputs -> bf16
  n = (int)W4;
  cvt_bf16<<<(n + 255) / 256, blk, 0, stream>>>(Wi_r, Wi_r_b, n);
  cvt_bf16<<<(n + 255) / 256, blk, 0, stream>>>(Wh_r, Wh_r_b, n);
  cvt_bf16<<<(n + 255) / 256, blk, 0, stream>>>(Wi_u, Wi_u_b, n);
  cvt_bf16<<<(n + 255) / 256, blk, 0, stream>>>(Wh_u, Wh_u_b, n);
  hipMemsetAsync(Wc_b, 0, (size_t)CPAD * Hh * 2, stream);   // zero pad rows 2513..2559
  n = Cc * Hh;
  cvt_bf16<<<(n + 255) / 256, blk, 0, stream>>>(Wc, Wc_b, n);
  n = Tt * Bb * Ff;
  cvt_x_tbf<<<(n + 255) / 256, blk, 0, stream>>>(inputs, X_b, n);
  n = 4 * Hh;
  add_vec<<<(n + 255) / 256, blk, 0, stream>>>(bi_r, bh_r, br, n);
  add_vec<<<(n + 255) / 256, blk, 0, stream>>>(bi_u, bh_u, bu, n);
  hipMemsetAsync(h0, 0, BH * 2, stream);
  hipMemsetAsync(c0, 0, BH * 4, stream);

  // input projections: [2048,1024] @ [1024,4096]^T(row-major [4096,1024]) + bias
  gemm_bf16_bias<<<dim3(32, 16), blk, 0, stream>>>(X_b, Wi_r_b, br, XI_r, 4 * Hh, Ff, 0);
  gemm_bf16_bias<<<dim3(32, 16), blk, 0, stream>>>(X_b, Wi_u_b, bu, XI_u, 4 * Hh, Ff, 0);

  // rolling LSTM: 16 sequential steps, collect (hs[t], cs[t])
  for (int t = 0; t < Tt; ++t) {
    lstm_step<<<dim3(Hh / 32, 1), blk, 0, stream>>>(
        Wh_r_b,
        t ? hs + (size_t)(t - 1) * BH : h0,
        XI_r + (size_t)t * Bb * 4 * Hh,
        t ? cs + (size_t)(t - 1) * BH : c0,
        cs + (size_t)t * BH,
        hs + (size_t)t * BH,
        (unsigned short*)nullptr, -1);
  }

  // unrolling LSTM: t-batched; sequence t needs (17 - t) steps.
  // step a (1..17): active t in [0, min(15, 17-a)]; t = 17-a does its final
  // step and writes its hidden into hf. h ping-pongs hs<->hp; c in place.
  for (int a = 1; a <= Tt + 1; ++a) {
    int cnt = (18 - a) < Tt ? (18 - a) : Tt;
    const unsigned short* src = (a & 1) ? hs : hp;
    unsigned short* dst = (a & 1) ? hp : hs;
    lstm_step<<<dim3(Hh / 32, cnt), blk, 0, stream>>>(
        Wh_u_b, src, XI_u, cs, cs, dst, hf, 17 - a);
  }

  // classifier: y[b,t,:] = hf[t,b,:] @ Wc^T + bc  (row remap t*B+b -> b*T+t)
  gemm_bf16_bias<<<dim3(CPAD / 128, 16), blk, 0, stream>>>(hf, Wc_b, bc, out, Cc, Hh, 1);
}
